// ElectrolyteGNN_3470333575795
// MI455X (gfx1250) — compile-verified, matrix operand order checked by reading, not run on hardware
//
#include <hip/hip_runtime.h>
#include <hip/hip_bf16.h>

typedef __attribute__((ext_vector_type(16))) _Float16 v16h;
typedef __attribute__((ext_vector_type(8)))  _Float16 v8h;
typedef __attribute__((ext_vector_type(4)))  _Float16 v4h;
typedef __attribute__((ext_vector_type(8)))  float    v8f;

#define GN      20000
#define GE      160000
#define GETOT   180000   // E + N self loops
#define GB      256
#define GIN     177
#define GK1P    192      // GIN padded to multiple of 32
#define GHC     512
#define GH      4
#define GC      128
#define NEG_SLOPE 0.2f

// ---------------------------------------------------------------------------
// helpers
// ---------------------------------------------------------------------------
__device__ __forceinline__ void atomicMaxFloat(float* addr, float value) {
    if (value >= 0.0f)
        atomicMax((int*)addr, __float_as_int(value));
    else
        atomicMin((unsigned int*)addr, __float_as_uint(value));
}

__device__ __forceinline__ float lrelu(float v) {
    return (v >= 0.0f) ? v : v * NEG_SLOPE;
}

// low 32 bits of a flat shared-memory pointer == LDS group-segment byte offset
__device__ __forceinline__ unsigned int lds_off(const void* p) {
    return (unsigned int)(unsigned long long)p;
}

// CDNA5 async global -> LDS copies (no VGPR staging, tracked by ASYNCcnt)
__device__ __forceinline__ void async_ld_b64(unsigned int dst, const void* src) {
    asm volatile("global_load_async_to_lds_b64 %0, %1, off"
                 :: "v"(dst), "v"((unsigned long long)(uintptr_t)src)
                 : "memory");
}
__device__ __forceinline__ void async_ld_b128(unsigned int dst, const void* src) {
    asm volatile("global_load_async_to_lds_b128 %0, %1, off"
                 :: "v"(dst), "v"((unsigned long long)(uintptr_t)src)
                 : "memory");
}

__global__ void fill_kernel(float* __restrict__ p, float v, int n) {
    int i = blockIdx.x * blockDim.x + threadIdx.x;
    if (i < n) p[i] = v;
}

// pad-convert activation: src [M x K] f32 -> dst [M x Kp] f16 (zero padded)
__global__ void cvt_pad_a_kernel(const float* __restrict__ src,
                                 _Float16* __restrict__ dst,
                                 int M, int K, int Kp) {
    int i = blockIdx.x * blockDim.x + threadIdx.x;
    if (i >= M * Kp) return;
    int m = i / Kp, k = i - m * Kp;
    dst[i] = (k < K) ? (_Float16)src[(size_t)m * K + k] : (_Float16)0.0f;
}

// transpose-convert weight: W [K x 512] f32 -> WT [512 x Kp] f16 (zero padded)
__global__ void cvt_wt_kernel(const float* __restrict__ W,
                              _Float16* __restrict__ WT,
                              int K, int Kp) {
    int i = blockIdx.x * blockDim.x + threadIdx.x;
    if (i >= GHC * Kp) return;
    int n = i / Kp, k = i - n * Kp;
    WT[i] = (k < K) ? (_Float16)W[(size_t)k * GHC + n] : (_Float16)0.0f;
}

// ---------------------------------------------------------------------------
// self-loop edge-attr mean (fill_value='mean')
// ---------------------------------------------------------------------------
__global__ void sl_accum_kernel(const int* __restrict__ edge_index,
                                const float* __restrict__ edge_attr,
                                float* __restrict__ easum, float* __restrict__ cnt) {
    int e = blockIdx.x * blockDim.x + threadIdx.x;
    if (e >= GE) return;
    int dst = edge_index[GE + e];
    atomicAdd(&easum[dst * 2 + 0], edge_attr[e * 2 + 0]);
    atomicAdd(&easum[dst * 2 + 1], edge_attr[e * 2 + 1]);
    atomicAdd(&cnt[dst], 1.0f);
}

__global__ void sl_final_kernel(const float* __restrict__ easum,
                                const float* __restrict__ cnt,
                                float* __restrict__ slea) {
    int n = blockIdx.x * blockDim.x + threadIdx.x;
    if (n >= GN) return;
    float c = fmaxf(cnt[n], 1.0f);
    slea[n * 2 + 0] = easum[n * 2 + 0] / c;
    slea[n * 2 + 1] = easum[n * 2 + 1] / c;
}

// ---------------------------------------------------------------------------
// Dual GEMM (f16 inputs, pre-padded/pre-transposed):
//   Xl = A@Wl + bl, Xr = A@Wr + br
//   A:   M x Kp   f16 row-major (Kp multiple of 32, M multiple of 32)
//   WT:  512 x Kp f16 n-major (pre-transposed)
// Block: 256 thr = 8 waves; tile 32(M) x 64(N); K-step 32.
// Double-buffered LDS fed by async global->LDS DMA; last tile peeled so the
// steady-state loop is branch-free.
// ---------------------------------------------------------------------------
__global__ __launch_bounds__(256) void gemm_dual_kernel(
    const _Float16* __restrict__ A,
    const _Float16* __restrict__ WlT, const float* __restrict__ bl,
    const _Float16* __restrict__ WrT, const float* __restrict__ br,
    float* __restrict__ Xl, float* __restrict__ Xr,
    int Kp)
{
    __shared__ __align__(16) _Float16 As[2][32 * 40];       // [m][k], stride 40
    __shared__ __align__(16) _Float16 Bs[2][2][64 * 40];    // [w][n][k], stride 40

    const int tid  = threadIdx.x;
    const int lane = tid & 31;
    const int wv   = tid >> 5;       // 0..7
    const int wm   = wv >> 2;        // 0..1 (M sub-tile)
    const int wn   = wv & 3;         // 0..3 (N sub-tile)

    const int row0 = blockIdx.x * 32;
    const int col0 = blockIdx.y * 64;

    // static per-thread load coordinates
    const int ar = tid >> 3;              // A row 0..31
    const int ak = (tid & 7) * 4;         // A k   0..28 step 4  (b64)
    const int bn = tid >> 2;              // B n   0..63
    const int bk = (tid & 3) * 8;         // B k   0..24 step 8  (b128)

    const _Float16* Ap  = A   + (size_t)(row0 + ar) * Kp + ak;
    const _Float16* Blp = WlT + (size_t)(col0 + bn) * Kp + bk;
    const _Float16* Brp = WrT + (size_t)(col0 + bn) * Kp + bk;

    const unsigned int aoff  = lds_off(&As[0][ar * 40 + ak]);
    const unsigned int b0off = lds_off(&Bs[0][0][bn * 40 + bk]);
    const unsigned int b1off = lds_off(&Bs[0][1][bn * 40 + bk]);
    const unsigned int abuf  = (unsigned int)(sizeof(_Float16) * 32 * 40);      // 2560
    const unsigned int bbuf  = (unsigned int)(sizeof(_Float16) * 2 * 64 * 40);  // 10240

    // per-wave frag addresses
    const int fr = (lane & 15) + wm * 16;
    const int fn = (lane & 15) + wn * 16;
    const int kb = (lane >> 4) * 8;

    v8f accL = {};
    v8f accR = {};

    auto compute_tile = [&](int cur) {
        const _Float16* as  = As[cur];
        const _Float16* bs0 = Bs[cur][0];
        const _Float16* bs1 = Bs[cur][1];

        v8h alo  = *(const v8h*)&as [fr * 40 + kb];
        v8h ahi  = *(const v8h*)&as [fr * 40 + kb + 16];
        v8h blo0 = *(const v8h*)&bs0[fn * 40 + kb];
        v8h bhi0 = *(const v8h*)&bs0[fn * 40 + kb + 16];
        v8h blo1 = *(const v8h*)&bs1[fn * 40 + kb];
        v8h bhi1 = *(const v8h*)&bs1[fn * 40 + kb + 16];

        v16h a, b0, b1;
        #pragma unroll
        for (int i = 0; i < 8; ++i) {
            a[i]  = alo[i];  a[i + 8]  = ahi[i];
            b0[i] = blo0[i]; b0[i + 8] = bhi0[i];
            b1[i] = blo1[i]; b1[i + 8] = bhi1[i];
        }

        accL = __builtin_amdgcn_wmma_f32_16x16x32_f16(
                   false, a, false, b0, (short)0, accL, false, false);
        accR = __builtin_amdgcn_wmma_f32_16x16x32_f16(
                   false, a, false, b1, (short)0, accR, false, false);
    };

    const int ksteps = Kp >> 5;

    // prologue: kick off tile 0 into buffer 0  (3 async instructions / wave)
    async_ld_b64 (aoff,  Ap);
    async_ld_b128(b0off, Blp);
    async_ld_b128(b1off, Brp);

    // steady state: branch-free; overlap next-tile DMA with current compute
    for (int kt = 0; kt < ksteps - 1; ++kt) {
        const int cur = kt & 1;
        const unsigned int nb = (unsigned int)(cur ^ 1);
        const int k0n = (kt + 1) * 32;

        async_ld_b64 (aoff  + nb * abuf, Ap  + k0n);
        async_ld_b128(b0off + nb * bbuf, Blp + k0n);
        async_ld_b128(b1off + nb * bbuf, Brp + k0n);
        asm volatile("s_wait_asynccnt 3" ::: "memory");  // current tile landed
        __syncthreads();

        compute_tile(cur);

        __syncthreads();   // protect buffer before it is refilled next round
    }

    // peeled last tile
    asm volatile("s_wait_asynccnt 0" ::: "memory");
    __syncthreads();
    compute_tile((ksteps - 1) & 1);

    // ---- epilogue (grid tiles M exactly; no bounds checks) ----
    const int n     = col0 + wn * 16 + (lane & 15);
    const int mbase = row0 + wm * 16 + (lane >> 4) * 8;
    const float bLv = bl[n];
    const float bRv = br[n];
    #pragma unroll
    for (int i = 0; i < 8; ++i) {
        int m = mbase + i;
        Xl[(size_t)m * GHC + n] = accL[i] + bLv;
        Xr[(size_t)m * GHC + n] = accR[i] + bRv;
    }
}

// ---------------------------------------------------------------------------
// Edge pass A: attention logits + segment max.  One wave32 per edge.
// ---------------------------------------------------------------------------
__global__ __launch_bounds__(256) void edge_alpha_kernel(
    const float* __restrict__ xl, const float* __restrict__ xr,
    const int* __restrict__ edge_index, const float* __restrict__ edge_attr,
    const float* __restrict__ slea, const float* __restrict__ We,
    const float* __restrict__ att, float* __restrict__ alpha,
    float* __restrict__ amax)
{
    int e    = (blockIdx.x * blockDim.x + threadIdx.x) >> 5;
    int lane = threadIdx.x & 31;
    if (e >= GETOT) return;

    int src, dst; float ea0, ea1;
    if (e < GE) {
        src = edge_index[e]; dst = edge_index[GE + e];
        ea0 = edge_attr[e * 2 + 0]; ea1 = edge_attr[e * 2 + 1];
    } else {
        src = dst = e - GE;
        ea0 = slea[src * 2 + 0]; ea1 = slea[src * 2 + 1];
    }

    const float* xls = xl + (size_t)src * GHC;
    const float* xrd = xr + (size_t)dst * GHC;
    const int c4 = lane * 4;

    float acc[GH];
    #pragma unroll
    for (int h = 0; h < GH; ++h) {
        const int base = h * GC + c4;
        float4 xa = *(const float4*)(xls + base);
        float4 xb = *(const float4*)(xrd + base);
        float4 w0 = *(const float4*)(We + base);
        float4 w1 = *(const float4*)(We + GHC + base);
        float4 at = *(const float4*)(att + base);
        float s;
        s  = lrelu(xa.x + xb.x + ea0 * w0.x + ea1 * w1.x) * at.x;
        s += lrelu(xa.y + xb.y + ea0 * w0.y + ea1 * w1.y) * at.y;
        s += lrelu(xa.z + xb.z + ea0 * w0.z + ea1 * w1.z) * at.z;
        s += lrelu(xa.w + xb.w + ea0 * w0.w + ea1 * w1.w) * at.w;
        acc[h] = s;
    }
    #pragma unroll
    for (int h = 0; h < GH; ++h) {
        float s = acc[h];
        #pragma unroll
        for (int o = 16; o > 0; o >>= 1) s += __shfl_xor(s, o, 32);
        acc[h] = s;
    }
    if (lane < GH) {
        alpha[(size_t)e * GH + lane] = acc[lane];
        atomicMaxFloat(&amax[dst * GH + lane], acc[lane]);
    }
}

// ---------------------------------------------------------------------------
// Edge pass B: exp(alpha - amax[dst]) + segment-sum denominator
// ---------------------------------------------------------------------------
__global__ void edge_exp_kernel(const int* __restrict__ edge_index,
                                float* __restrict__ alpha,
                                const float* __restrict__ amax,
                                float* __restrict__ denom)
{
    int i = blockIdx.x * blockDim.x + threadIdx.x;
    if (i >= GETOT * GH) return;
    int e = i >> 2, h = i & 3;
    int dst = (e < GE) ? edge_index[GE + e] : (e - GE);
    float ex = __expf(alpha[i] - amax[dst * GH + h]);
    alpha[i] = ex;
    atomicAdd(&denom[dst * GH + h], ex);
}

// ---------------------------------------------------------------------------
// Edge pass C: out[dst] += xl[src] * a   (wave32 per edge)
// ---------------------------------------------------------------------------
__global__ __launch_bounds__(256) void edge_aggr_kernel(
    const float* __restrict__ xl, const int* __restrict__ edge_index,
    const float* __restrict__ alpha, const float* __restrict__ denom,
    float* __restrict__ out)
{
    int e    = (blockIdx.x * blockDim.x + threadIdx.x) >> 5;
    int lane = threadIdx.x & 31;
    if (e >= GETOT) return;

    int src, dst;
    if (e < GE) { src = edge_index[e]; dst = edge_index[GE + e]; }
    else        { src = dst = e - GE; }

    float a[GH];
    #pragma unroll
    for (int h = 0; h < GH; ++h)
        a[h] = alpha[(size_t)e * GH + h] / (denom[dst * GH + h] + 1e-16f);

    const float* xls = xl + (size_t)src * GHC;
    float*       od  = out + (size_t)dst * GHC;
    const int c4 = lane * 4;
    #pragma unroll
    for (int h = 0; h < GH; ++h) {
        const int base = h * GC + c4;
        float4 xv = *(const float4*)(xls + base);
        float ah = a[h];
        atomicAdd(&od[base + 0], xv.x * ah);
        atomicAdd(&od[base + 1], xv.y * ah);
        atomicAdd(&od[base + 2], xv.z * ah);
        atomicAdd(&od[base + 3], xv.w * ah);
    }
}

// ---------------------------------------------------------------------------
// bias + optional ReLU (in place); optionally emit f16 copy for next GEMM
// ---------------------------------------------------------------------------
__global__ void bias_act_kernel(float* __restrict__ h,
                                const float* __restrict__ bo,
                                _Float16* __restrict__ h16,  // may be null
                                int total, int relu)
{
    int i = blockIdx.x * blockDim.x + threadIdx.x;
    if (i >= total) return;
    float v = h[i] + bo[i & (GHC - 1)];
    if (relu) v = fmaxf(v, 0.0f);
    h[i] = v;
    if (h16) h16[i] = (_Float16)v;
}

// ---------------------------------------------------------------------------
// global mean pool (sum + count via atomics)
// ---------------------------------------------------------------------------
__global__ void pool_kernel(const float* __restrict__ h,
                            const int* __restrict__ batch,
                            float* __restrict__ gsum, float* __restrict__ gcnt)
{
    int i = blockIdx.x * blockDim.x + threadIdx.x;
    if (i >= GN * GHC) return;
    int n = i >> 9, c = i & (GHC - 1);
    int b = batch[n];
    atomicAdd(&gsum[(size_t)b * GHC + c], h[i]);
    if (c == 0) atomicAdd(&gcnt[b], 1.0f);
}

// ---------------------------------------------------------------------------
// MLP head: relu(g @ W3 + b3) @ W4 + b4 ; one block (128 thr) per graph
// ---------------------------------------------------------------------------
__global__ __launch_bounds__(128) void mlp_kernel(
    const float* __restrict__ gsum, const float* __restrict__ gcnt,
    const float* __restrict__ W3, const float* __restrict__ b3,
    const float* __restrict__ W4, const float* __restrict__ b4,
    float* __restrict__ out)
{
    int b = blockIdx.x;
    int j = threadIdx.x;                  // hidden unit 0..127
    float inv = 1.0f / fmaxf(gcnt[b], 1.0f);
    const float* g = gsum + (size_t)b * GHC;

    float acc = b3[j];
    for (int k = 0; k < GHC; ++k)
        acc += (g[k] * inv) * W3[(size_t)k * GC + j];
    acc = fmaxf(acc, 0.0f);

    __shared__ float red[GC];
    red[j] = acc * W4[j];
    __syncthreads();
    for (int s = GC / 2; s > 0; s >>= 1) {
        if (j < s) red[j] += red[j + s];
        __syncthreads();
    }
    if (j == 0) out[b] = red[0] + b4[0];
}

// ---------------------------------------------------------------------------
// launch
// ---------------------------------------------------------------------------
extern "C" void kernel_launch(void* const* d_in, const int* in_sizes, int n_in,
                              void* d_out, int out_size, void* d_ws, size_t ws_size,
                              hipStream_t stream) {
    const float* x    = (const float*)d_in[0];
    const int*   ei   = (const int*)  d_in[1];
    const float* ea   = (const float*)d_in[2];
    const int*   bat  = (const int*)  d_in[3];
    const float* Wl1  = (const float*)d_in[4];
    const float* bl1  = (const float*)d_in[5];
    const float* Wr1  = (const float*)d_in[6];
    const float* br1  = (const float*)d_in[7];
    const float* We1  = (const float*)d_in[8];
    const float* att1 = (const float*)d_in[9];
    const float* bo1  = (const float*)d_in[10];
    const float* Wl2  = (const float*)d_in[11];
    const float* bl2  = (const float*)d_in[12];
    const float* Wr2  = (const float*)d_in[13];
    const float* br2  = (const float*)d_in[14];
    const float* We2  = (const float*)d_in[15];
    const float* att2 = (const float*)d_in[16];
    const float* bo2  = (const float*)d_in[17];
    const float* W3   = (const float*)d_in[18];
    const float* b3   = (const float*)d_in[19];
    const float* W4   = (const float*)d_in[20];
    const float* b4   = (const float*)d_in[21];
    float* out = (float*)d_out;

    // ---- workspace layout (float units, 64B aligned blocks) ----
    float* ws = (float*)d_ws;
    size_t o = 0;
    auto alloc = [&](size_t nfloats) {
        float* p = ws + o; o += (nfloats + 15) & ~(size_t)15; return p;
    };
    auto alloch = [&](size_t nhalf) { return (_Float16*)alloc((nhalf + 1) / 2); };

    float* xl    = alloc((size_t)GN * GHC);
    float* xr    = alloc((size_t)GN * GHC);
    float* hbuf  = alloc((size_t)GN * GHC);     // h1, later reused for h2
    float* alpha = alloc((size_t)GETOT * GH);
    float* amax  = alloc((size_t)GN * GH);
    float* denom = alloc((size_t)GN * GH);
    float* slea  = alloc((size_t)GN * 2);
    float* easum = alloc((size_t)GN * 2);
    float* cnt   = alloc((size_t)GN);
    float* gsum  = alloc((size_t)GB * GHC);
    float* gcnt  = alloc((size_t)GB);
    _Float16* xh   = alloch((size_t)GN * GK1P);   // padded f16 x
    _Float16* ah   = alloch((size_t)GN * GHC);    // f16 relu(h1)
    _Float16* WlT1 = alloch((size_t)GHC * GK1P);
    _Float16* WrT1 = alloch((size_t)GHC * GK1P);
    _Float16* WlT2 = alloch((size_t)GHC * GHC);
    _Float16* WrT2 = alloch((size_t)GHC * GHC);

    const int T = 256;
    auto blk = [](long n, int t) { return (int)((n + t - 1) / t); };

    dim3 gemm_grid(GN / 32, GHC / 64);     // 625 x 8
    const int edge_blocks = blk((long)GETOT * 32, T);

    // ---- precision prep: pad/convert activations + transpose/convert weights ----
    cvt_pad_a_kernel<<<blk((long)GN * GK1P, T), T, 0, stream>>>(x, xh, GN, GIN, GK1P);
    cvt_wt_kernel<<<blk((long)GHC * GK1P, T), T, 0, stream>>>(Wl1, WlT1, GIN, GK1P);
    cvt_wt_kernel<<<blk((long)GHC * GK1P, T), T, 0, stream>>>(Wr1, WrT1, GIN, GK1P);
    cvt_wt_kernel<<<blk((long)GHC * GHC, T), T, 0, stream>>>(Wl2, WlT2, GHC, GHC);
    cvt_wt_kernel<<<blk((long)GHC * GHC, T), T, 0, stream>>>(Wr2, WrT2, GHC, GHC);

    // ---- self-loop mean edge attrs ----
    fill_kernel<<<blk(GN * 2, T), T, 0, stream>>>(easum, 0.0f, GN * 2);
    fill_kernel<<<blk(GN, T), T, 0, stream>>>(cnt, 0.0f, GN);
    sl_accum_kernel<<<blk(GE, T), T, 0, stream>>>(ei, ea, easum, cnt);
    sl_final_kernel<<<blk(GN, T), T, 0, stream>>>(easum, cnt, slea);

    // ======== GATv2 layer 1 ========
    gemm_dual_kernel<<<gemm_grid, T, 0, stream>>>(xh, WlT1, bl1, WrT1, br1, xl, xr, GK1P);
    fill_kernel<<<blk(GN * GH, T), T, 0, stream>>>(amax, -1e30f, GN * GH);
    fill_kernel<<<blk(GN * GH, T), T, 0, stream>>>(denom, 0.0f, GN * GH);
    edge_alpha_kernel<<<edge_blocks, T, 0, stream>>>(xl, xr, ei, ea, slea, We1, att1, alpha, amax);
    edge_exp_kernel<<<blk((long)GETOT * GH, T), T, 0, stream>>>(ei, alpha, amax, denom);
    fill_kernel<<<blk((long)GN * GHC, T), T, 0, stream>>>(hbuf, 0.0f, GN * GHC);
    edge_aggr_kernel<<<edge_blocks, T, 0, stream>>>(xl, ei, alpha, denom, hbuf);
    bias_act_kernel<<<blk((long)GN * GHC, T), T, 0, stream>>>(hbuf, bo1, ah, GN * GHC, 1);

    // ======== GATv2 layer 2 ========
    gemm_dual_kernel<<<gemm_grid, T, 0, stream>>>(ah, WlT2, bl2, WrT2, br2, xl, xr, GHC);
    fill_kernel<<<blk(GN * GH, T), T, 0, stream>>>(amax, -1e30f, GN * GH);
    fill_kernel<<<blk(GN * GH, T), T, 0, stream>>>(denom, 0.0f, GN * GH);
    edge_alpha_kernel<<<edge_blocks, T, 0, stream>>>(xl, xr, ei, ea, slea, We2, att2, alpha, amax);
    edge_exp_kernel<<<blk((long)GETOT * GH, T), T, 0, stream>>>(ei, alpha, amax, denom);
    fill_kernel<<<blk((long)GN * GHC, T), T, 0, stream>>>(hbuf, 0.0f, GN * GHC);
    edge_aggr_kernel<<<edge_blocks, T, 0, stream>>>(xl, ei, alpha, denom, hbuf);
    bias_act_kernel<<<blk((long)GN * GHC, T), T, 0, stream>>>(hbuf, bo2, (_Float16*)nullptr, GN * GHC, 0);

    // ======== pool + MLP ========
    fill_kernel<<<blk(GB * GHC, T), T, 0, stream>>>(gsum, 0.0f, GB * GHC);
    fill_kernel<<<blk(GB, T), T, 0, stream>>>(gcnt, 0.0f, GB);
    pool_kernel<<<blk((long)GN * GHC, T), T, 0, stream>>>(hbuf, bat, gsum, gcnt);
    mlp_kernel<<<GB, GC, 0, stream>>>(gsum, gcnt, W3, b3, W4, b4, out);
}
